// Cross_Attention_32469952757796
// MI455X (gfx1250) — matrix operand order repaired and checked by reading, not compiled, not run on hardware
//
#include <hip/hip_runtime.h>

typedef __attribute__((ext_vector_type(16))) _Float16     v16h;
typedef __attribute__((ext_vector_type(8)))  float        v8f;
typedef __attribute__((ext_vector_type(2)))  unsigned int u32x2;
typedef __attribute__((ext_vector_type(4)))  unsigned int u32x4;

#define NB      8
#define HEADS   8
#define HC      16                 // channels per head
#define NN      (128*128)          // 16384 positions
#define BH      (NB*HEADS)         // 64
#define THREADS 256

// ---------------- kernel 0: zero the context accumulator ----------------
__global__ void zero_ctx_kernel(float* __restrict__ ctx) {
    int i = blockIdx.x * blockDim.x + threadIdx.x;
    if (i < BH * HC * HC) ctx[i] = 0.0f;
}

// ---------- kernel 1: per-(bh,kc) row max and 1/sum(exp) over N ---------
__global__ void row_stats_kernel(const float* __restrict__ x2,
                                 float* __restrict__ rowmax,
                                 float* __restrict__ rowinv) {
    const int row = blockIdx.x;                       // 0 .. BH*HC-1
    const float4* p = (const float4*)(x2 + (size_t)row * NN);
    const int tid = threadIdx.x;
    __shared__ float red[THREADS];

    float m = -3.4e38f;
    for (int i = tid; i < NN / 4; i += THREADS) {
        float4 v = p[i];
        m = fmaxf(m, fmaxf(fmaxf(v.x, v.y), fmaxf(v.z, v.w)));
    }
    red[tid] = m; __syncthreads();
    for (int s = THREADS / 2; s > 0; s >>= 1) {
        if (tid < s) red[tid] = fmaxf(red[tid], red[tid + s]);
        __syncthreads();
    }
    m = red[0];
    __syncthreads();

    float sum = 0.f;
    for (int i = tid; i < NN / 4; i += THREADS) {
        float4 v = p[i];
        sum += __expf(v.x - m) + __expf(v.y - m) + __expf(v.z - m) + __expf(v.w - m);
    }
    red[tid] = sum; __syncthreads();
    for (int s = THREADS / 2; s > 0; s >>= 1) {
        if (tid < s) red[tid] += red[tid + s];
        __syncthreads();
    }
    if (tid == 0) { rowmax[row] = m; rowinv[row] = 1.0f / red[0]; }
}

// ------ kernel 2: context_raw[bh,kc,vc] = sum_n exp(k-max) * v (WMMA) ----
#define SPLIT 8
#define CHUNK 512

__global__ void context_kernel(const float* __restrict__ x1,   // values
                               const float* __restrict__ x2,   // keys
                               const float* __restrict__ rowmax,
                               float* __restrict__ ctx) {
    const int bh   = blockIdx.x / SPLIT;
    const int part = blockIdx.x % SPLIT;
    const int n0   = part * (NN / SPLIT);             // 2048-wide N slice
    const int tid  = threadIdx.x;
    const int lane = tid & 31;
    const int wv   = tid >> 5;                        // 8 waves
    const int m    = lane & 15;
    const int hi   = lane >> 4;                       // lane half (ISA A/B layout)

    __shared__ __align__(16) _Float16 lkh[HC * CHUNK]; // exp(k - rowmax), f16
    __shared__ __align__(16) _Float16 lvh[HC * CHUNK]; // v, f16
    __shared__ float smax[HC];

    if (tid < HC) smax[tid] = rowmax[bh * HC + tid];

    v8f acc = {};

    for (int cb = n0; cb < n0 + NN / SPLIT; cb += CHUNK) {
        const bool more = (cb + CHUNK) < (n0 + NN / SPLIT);
        __syncthreads();
        // float4 global loads, convert to f16 once, 8-byte LDS stores
        for (int i = tid; i < HC * CHUNK / 4; i += THREADS) {
            const int r = i / (CHUNK / 4);
            const int c = (i - r * (CHUNK / 4)) * 4;
            const size_t g = (size_t)(bh * HC + r) * NN + cb + c;
            const float4 kf = *(const float4*)(x2 + g);
            const float4 vf = *(const float4*)(x1 + g);
            if (more) {
                __builtin_prefetch(x2 + g + CHUNK, 0, 0);
                __builtin_prefetch(x1 + g + CHUNK, 0, 0);
            }
            const float mr = smax[r];
            union { _Float16 h[4]; u32x2 u; } pk, pv;
            pk.h[0] = (_Float16)__expf(kf.x - mr);
            pk.h[1] = (_Float16)__expf(kf.y - mr);
            pk.h[2] = (_Float16)__expf(kf.z - mr);
            pk.h[3] = (_Float16)__expf(kf.w - mr);
            pv.h[0] = (_Float16)vf.x;  pv.h[1] = (_Float16)vf.y;
            pv.h[2] = (_Float16)vf.z;  pv.h[3] = (_Float16)vf.w;
            *(u32x2*)(lkh + r * CHUNK + c) = pk.u;
            *(u32x2*)(lvh + r * CHUNK + c) = pv.u;
        }
        __syncthreads();

        // Each wave owns a 64-wide slice -> two K=32 WMMA slabs per chunk.
        const int nb = wv * (CHUNK / 8);
        #pragma unroll
        for (int s = 0; s < CHUNK / 8; s += 32) {
            const int ka = nb + s + hi * 8;   // A: lanes<16 K0-7/16-23, lanes>=16 K8-15/24-31
            const int kb = nb + s + hi * 8;   // B: lanes<16 K0-7/16-23, lanes>=16 K8-15/24-31
            union { v16h v; u32x4 q[2]; } a, b;
            a.q[0] = *(const u32x4*)(lkh + m * CHUNK + ka);        // halves 0-7
            a.q[1] = *(const u32x4*)(lkh + m * CHUNK + ka + 16);   // halves 8-15
            b.q[0] = *(const u32x4*)(lvh + m * CHUNK + kb);
            b.q[1] = *(const u32x4*)(lvh + m * CHUNK + kb + 16);
            acc = __builtin_amdgcn_wmma_f32_16x16x32_f16(false, a.v, false, b.v,
                                                         (short)0, acc, false, false);
        }
    }

    // D layout: acc[r] is (M = r + hi*8, col = lane&15). Merge partials in L2.
    float* cp = ctx + bh * (HC * HC);
    #pragma unroll
    for (int r = 0; r < 8; ++r)
        atomicAdd(&cp[(r + hi * 8) * HC + m], acc[r]);
}

// -- kernel 3: attended[vc,n] = sum_kc context[kc,vc] * query_sm[kc,n] ----
#define OSPLIT 16
#define CH3    128

__global__ void attend_kernel(const float* __restrict__ x2,
                              const float* __restrict__ rowinv,
                              const float* __restrict__ ctxraw,
                              float* __restrict__ out) {
    const int bh   = blockIdx.x / OSPLIT;
    const int part = blockIdx.x % OSPLIT;
    const int n0   = part * (NN / OSPLIT);            // 1024 columns
    const int tid  = threadIdx.x;
    const int lane = tid & 31;
    const int wv   = tid >> 5;
    const int m    = lane & 15;
    const int hi   = lane >> 4;

    __shared__ __align__(16) float lk[HC * CH3];
    __shared__ float sctx[HC][HC];

    {   // finalize context: scale row kc by 1/sum(exp) of the key softmax
        int kc = tid >> 4, vc = tid & 15;
        sctx[kc][vc] = ctxraw[bh * (HC * HC) + tid] * rowinv[bh * HC + kc];
    }
    __syncthreads();

    // A = context^T (M = vc, K = kc), K padded 16..31 with zeros.
    v16h a = {};
    #pragma unroll
    for (int e = 0; e < 16; ++e)
        if (hi == 0) a[e] = (_Float16)sctx[e][m];

    for (int cb = n0; cb < n0 + NN / OSPLIT; cb += CH3) {
        const bool more = (cb + CH3) < (n0 + NN / OSPLIT);
        __syncthreads();
        for (int i = tid; i < HC * CH3 / 4; i += THREADS) {     // float4 staging
            const int r = i / (CH3 / 4);
            const int c = (i - r * (CH3 / 4)) * 4;
            const size_t g = (size_t)(bh * HC + r) * NN + cb + c;
            *(float4*)(lk + r * CH3 + c) = *(const float4*)(x2 + g);
            if (more) __builtin_prefetch(x2 + g + CH3, 0, 0);
        }
        __syncthreads();

        const int c0 = wv * 16;        // this wave's 16 output columns
        v16h b = {};                   // K rows 16..31 stay zero (lanes 16-31)
        if (hi == 0) {
            float q[HC];
            float mx = -3.4e38f;
            #pragma unroll
            for (int kc = 0; kc < HC; ++kc) {
                q[kc] = lk[kc * CH3 + c0 + m];
                mx = fmaxf(mx, q[kc]);
            }
            float s = 0.f;
            #pragma unroll
            for (int kc = 0; kc < HC; ++kc) { q[kc] = __expf(q[kc] - mx); s += q[kc]; }
            const float inv = 1.0f / s;
            #pragma unroll
            for (int e = 0; e < HC; ++e) b[e] = (_Float16)(q[e] * inv);
        }
        v8f d = {};
        d = __builtin_amdgcn_wmma_f32_16x16x32_f16(false, a, false, b,
                                                   (short)0, d, false, false);
        const int col = cb + c0 + m;
        #pragma unroll
        for (int r = 0; r < 8; ++r)     // two coalesced 64B segments per store
            out[(size_t)(bh * HC + r + hi * 8) * NN + col] = d[r];
    }
}

extern "C" void kernel_launch(void* const* d_in, const int* in_sizes, int n_in,
                              void* d_out, int out_size, void* d_ws, size_t ws_size,
                              hipStream_t stream) {
    const float* x1 = (const float*)d_in[0];   // values
    const float* x2 = (const float*)d_in[1];   // keys / queries
    float* out = (float*)d_out;

    float* rowmax = (float*)d_ws;              // BH*HC
    float* rowinv = rowmax + BH * HC;          // BH*HC
    float* ctx    = rowinv + BH * HC;          // BH*HC*HC accumulator

    zero_ctx_kernel<<<(BH * HC * HC + THREADS - 1) / THREADS, THREADS, 0, stream>>>(ctx);
    row_stats_kernel<<<BH * HC, THREADS, 0, stream>>>(x2, rowmax, rowinv);
    context_kernel<<<BH * SPLIT, THREADS, 0, stream>>>(x1, x2, rowmax, ctx);
    attend_kernel<<<BH * OSPLIT, THREADS, 0, stream>>>(x2, rowinv, ctx, out);
}
